// LuongAttentionLayer_29489245454493
// MI455X (gfx1250) — matrix operand order repaired and checked by reading, not compiled, not run on hardware
//
#include <hip/hip_runtime.h>

// ---------------------------------------------------------------------------
// Fused Luong attention for MI455X (gfx1250, wave32, WMMA + async-to-LDS).
//   score[b,e,t] = enc[b,e,:] . dec[b,t,:]
//   attn = softmax over e;  ctx[b,t,:] = sum_e attn[e,t] * enc[b,e,:]
//   out  = concat([dec, ctx], -1)
//
// Fast path (needs 16 MiB workspace):
//   pass 1: convert enc f32 -> f16 row-major (ench) and f16 transposed (enct)
//   pass 2: flash-attention main loop; enc tiles staged into double-buffered
//           LDS with global_load_async_to_lds_b128 (ASYNCcnt), overlapped
//           with WMMA compute on the previous tile.
// Slow path (tiny/absent workspace): self-contained kernel converting inline.
// ---------------------------------------------------------------------------

typedef __attribute__((ext_vector_type(16))) _Float16     v16h;
typedef __attribute__((ext_vector_type(8)))  float        v8f;
typedef __attribute__((ext_vector_type(4)))  float        f32x4;
typedef __attribute__((ext_vector_type(4)))  int          v4i;
typedef __attribute__((ext_vector_type(4)))  unsigned int u32x4;
typedef __attribute__((ext_vector_type(2)))  unsigned int u32x2;

constexpr int Bb = 8;     // batch
constexpr int Te = 2048;  // encoder time
constexpr int Td = 2048;  // decoder time
constexpr int Dm = 256;   // model dim
constexpr int BQ = 64;    // t rows per workgroup (16 per wave)
constexpr int BK = 32;    // e rows per LDS tile / iteration
constexpr int QP = 264;   // q_lds row pitch (f16 elems)
constexpr int RP = 264;   // enc row-major pitch (f16 elems)
constexpr int TP = 40;    // enc transposed pitch (f16 elems)
constexpr int LDS_Q = BQ * QP;
constexpr int LDS_R = BK * RP;
constexpr int LDS_T = Dm * TP;
constexpr size_t LDS_BYTES_SLOW = (size_t)(LDS_Q + LDS_R + LDS_T) * 2;
constexpr size_t LDS_BYTES_FAST = (size_t)(LDS_Q + 2 * LDS_R + 2 * LDS_T) * 2;
constexpr size_t WS_NEEDED = 2ull * Bb * Te * Dm * sizeof(_Float16); // ench+enct

union H4 { _Float16 h[4]; u32x2 u; };

// Load a 16-bit WMMA operand as two 16-byte runs from LDS.
__device__ __forceinline__ v16h ld2(const _Float16* p0, const _Float16* p1) {
    union { u32x4 q[2]; v16h v; } u;
    u.q[0] = *(const u32x4*)p0;
    u.q[1] = *(const u32x4*)p1;
    return u.v;
}

// ---- gfx1250 async global->LDS copy (ASYNCcnt) with sync fallback ----------
#if __has_builtin(__builtin_amdgcn_global_load_async_to_lds_b128) && \
    __has_builtin(__builtin_amdgcn_s_wait_asynccnt)
#define HAVE_ASYNC 1
__device__ __forceinline__ void cp16(void* l, const void* g) {
    __builtin_amdgcn_global_load_async_to_lds_b128(
        (__attribute__((address_space(1))) v4i*)g,
        (__attribute__((address_space(3))) v4i*)l, 0, 0);
}
__device__ __forceinline__ void cp_wait() {
    __builtin_amdgcn_s_wait_asynccnt(0);
}
#else
#define HAVE_ASYNC 0
__device__ __forceinline__ void cp16(void* l, const void* g) {
    *(u32x4*)l = *(const u32x4*)g;
}
__device__ __forceinline__ void cp_wait() {}
#endif

// ---------------------------------------------------------------------------
// Pass 1: enc f32 -> ench (f16 row-major [b][e][d]) + enct (f16 transposed
// [b][d][e]). 64x64 tiles through LDS; all global accesses coalesced.
// ---------------------------------------------------------------------------
__global__ void __launch_bounds__(256)
convert_enc_kernel(const float* __restrict__ enc,
                   _Float16* __restrict__ ench,
                   _Float16* __restrict__ enct)
{
    __shared__ _Float16 tile[64][72];   // +8 f16 pad vs bank conflicts

    const int tx = threadIdx.x;
    const int r  = tx >> 4;             // 0..15
    const int c4 = (tx & 15) * 4;       // 0..60

    const size_t ebase = ((size_t)blockIdx.z * Te + (size_t)blockIdx.y * 64);
    const float*    src   = enc  + ebase * Dm + blockIdx.x * 64;
    _Float16*       dst_h = ench + ebase * Dm + blockIdx.x * 64;

#pragma unroll
    for (int k = 0; k < 4; ++k) {
        const int row = r + 16 * k;
        f32x4 v = *(const f32x4*)(src + (size_t)row * Dm + c4);
        H4 pk;
        pk.h[0] = (_Float16)v.x; pk.h[1] = (_Float16)v.y;
        pk.h[2] = (_Float16)v.z; pk.h[3] = (_Float16)v.w;
        *(u32x2*)(dst_h + (size_t)row * Dm + c4) = pk.u;
        *(u32x2*)(&tile[row][c4]) = pk.u;
    }
    __syncthreads();

    _Float16* dst_t = enct + ((size_t)blockIdx.z * Dm + (size_t)blockIdx.x * 64) * Te
                           + (size_t)blockIdx.y * 64;
#pragma unroll
    for (int k = 0; k < 4; ++k) {
        const int drow = r + 16 * k;
        H4 pk;
        pk.h[0] = tile[c4 + 0][drow];
        pk.h[1] = tile[c4 + 1][drow];
        pk.h[2] = tile[c4 + 2][drow];
        pk.h[3] = tile[c4 + 3][drow];
        *(u32x2*)(dst_t + (size_t)drow * Te + c4) = pk.u;
    }
}

// ---------------------------------------------------------------------------
// Shared flash-attention compute core for one 32-e tile.
// Per wave: 16 t-rows (t = lane&15), online softmax in registers, ctx^T acc.
// ---------------------------------------------------------------------------
struct WaveState {
    v8f   acc[16];
    float m, l;
};

__device__ __forceinline__ void attn_tile(WaveState& st,
                                          const _Float16* q_lds,
                                          const _Float16* erow,
                                          const _Float16* etr,
                                          int l16, int half, int tql)
{
    // ---- S^T = enc(16e x 32d tiles) x Q^T(32d x 16t), f32 accum.
    v8f cs[2];
#pragma unroll
    for (int s = 0; s < 2; ++s)
#pragma unroll
        for (int v = 0; v < 8; ++v) cs[s][v] = 0.0f;

#pragma unroll
    for (int k = 0; k < Dm / 32; ++k) {
        const _Float16* qb = q_lds + tql * QP + k * 32 + 16 * half;
        v16h bq = ld2(qb, qb + 8);
#pragma unroll
        for (int s = 0; s < 2; ++s) {
            const _Float16* ar = erow + (s * 16 + l16) * RP + k * 32 + 16 * half;
            v16h ae = ld2(ar, ar + 8);
            cs[s] = __builtin_amdgcn_wmma_f32_16x16x32_f16(
                        false, ae, false, bq, (short)0, cs[s], false, false);
        }
    }

    // ---- Online softmax over this 32-e block (per lane: row t = l16)
    float bm = -1e30f;
#pragma unroll
    for (int s = 0; s < 2; ++s)
#pragma unroll
        for (int v = 0; v < 8; ++v) bm = fmaxf(bm, cs[s][v]);
    bm = fmaxf(bm, __shfl_xor(bm, 16));

    const float mn    = fmaxf(st.m, bm);
    const float alpha = __expf(st.m - mn);

    float ps0[8], ps1[8];
    float rs = 0.0f;
#pragma unroll
    for (int v = 0; v < 8; ++v) { ps0[v] = __expf(cs[0][v] - mn); rs += ps0[v]; }
#pragma unroll
    for (int v = 0; v < 8; ++v) { ps1[v] = __expf(cs[1][v] - mn); rs += ps1[v]; }
    rs += __shfl_xor(rs, 16);
    st.l = st.l * alpha + rs;

    if (__any(mn > st.m)) {
#pragma unroll
        for (int i = 0; i < 16; ++i)
#pragma unroll
            for (int v = 0; v < 8; ++v) st.acc[i][v] *= alpha;
    }
    st.m = mn;

    // ---- Repack P from C layout into B layout (K = e = r + 16*half)
    float o0[8], o1[8];
#pragma unroll
    for (int v = 0; v < 8; ++v) {
        o0[v] = __shfl_xor(ps0[v], 16);
        o1[v] = __shfl_xor(ps1[v], 16);
    }
    v16h bp;
#pragma unroll
    for (int r = 0; r < 8; ++r) bp[r]     = (_Float16)(half ? o1[r]  : ps0[r]);
#pragma unroll
    for (int r = 0; r < 8; ++r) bp[8 + r] = (_Float16)(half ? ps1[r] : o0[r]);

    // ---- ctx^T[d,t] += enc^T(16d x 32e) x P(32e x 16t)
#pragma unroll
    for (int dt = 0; dt < 16; ++dt) {
        const _Float16* ap = etr + (dt * 16 + l16) * TP + 16 * half;
        v16h ea = ld2(ap, ap + 8);
        st.acc[dt] = __builtin_amdgcn_wmma_f32_16x16x32_f16(
                         false, ea, false, bp, (short)0, st.acc[dt], false, false);
    }
}

__device__ __forceinline__ void stage_q_and_out(const float* decB, float* outB,
                                                _Float16* q_lds, int t0, int tid)
{
    const int r  = tid >> 1;
    const int d0 = (tid & 1) * 128;
    const float* src = decB + (size_t)(t0 + r) * Dm + d0;
    float*       dst = outB + (size_t)(t0 + r) * (2 * Dm) + d0;
    _Float16*    qd  = q_lds + r * QP + d0;
#pragma unroll
    for (int i = 0; i < 32; ++i) {
        f32x4 v4 = ((const f32x4*)src)[i];
        ((f32x4*)dst)[i] = v4;
        H4 pk;
        pk.h[0] = (_Float16)v4.x; pk.h[1] = (_Float16)v4.y;
        pk.h[2] = (_Float16)v4.z; pk.h[3] = (_Float16)v4.w;
        *(u32x2*)(qd + i * 4) = pk.u;
    }
}

__device__ __forceinline__ void write_ctx(const WaveState& st, float* outB,
                                          int tg, int half)
{
    const float inv_l = 1.0f / st.l;
#pragma unroll
    for (int dt = 0; dt < 16; ++dt) {
        f32x4 v0, v1;
        v0.x = st.acc[dt][0] * inv_l; v0.y = st.acc[dt][1] * inv_l;
        v0.z = st.acc[dt][2] * inv_l; v0.w = st.acc[dt][3] * inv_l;
        v1.x = st.acc[dt][4] * inv_l; v1.y = st.acc[dt][5] * inv_l;
        v1.z = st.acc[dt][6] * inv_l; v1.w = st.acc[dt][7] * inv_l;
        float* dst = outB + (size_t)tg * (2 * Dm) + Dm + dt * 16 + 8 * half;
        ((f32x4*)dst)[0] = v0;
        ((f32x4*)dst)[1] = v1;
    }
}

// ---------------------------------------------------------------------------
// Pass 2 (fast): pre-converted f16 enc, async double-buffered LDS staging.
// ---------------------------------------------------------------------------
__device__ __forceinline__ void stage_tile_async(const _Float16* ench_b,
                                                 const _Float16* enct_b,
                                                 _Float16* erow, _Float16* etr,
                                                 int e0, int tid)
{
#pragma unroll
    for (int k = 0; k < 8; ++k) {               // erow: 32 rows x 256 f16
        const int c = tid + k * 128;
        const int row = c >> 5, cc = c & 31;
        cp16(erow + row * RP + cc * 8,
             ench_b + (size_t)(e0 + row) * Dm + cc * 8);
    }
#pragma unroll
    for (int k = 0; k < 8; ++k) {               // etr: 256 rows x 32 f16
        const int c = tid + k * 128;
        const int d = c >> 2, cc = c & 3;
        cp16(etr + d * TP + cc * 8,
             enct_b + (size_t)d * Te + e0 + cc * 8);
    }
}

__global__ void __launch_bounds__(128)
luong_attn_fast(const _Float16* __restrict__ ench,
                const _Float16* __restrict__ enct,
                const float* __restrict__ dec,
                float* __restrict__ out)
{
    extern __shared__ _Float16 smem[];
    _Float16* q_lds = smem;                       // [BQ][QP]
    _Float16* erow0 = smem + LDS_Q;               // 2 x [BK][RP]
    _Float16* etr0  = smem + LDS_Q + 2 * LDS_R;   // 2 x [Dm][TP]

    const int tid  = threadIdx.x;
    const int lane = tid & 31;
    const int wave = tid >> 5;
    const int half = lane >> 4;
    const int l16  = lane & 15;

    const int wg = blockIdx.x;
    const int b  = wg / (Td / BQ);
    const int t0 = (wg % (Td / BQ)) * BQ;

    const _Float16* enchB = ench + (size_t)b * Te * Dm;
    const _Float16* enctB = enct + (size_t)b * Dm * Te;
    const float*    decB  = dec  + (size_t)b * Td * Dm;
    float*          outB  = out  + (size_t)b * Td * (2 * Dm);

    stage_q_and_out(decB, outB, q_lds, t0, tid);
    stage_tile_async(enchB, enctB, erow0, etr0, 0, tid);
    cp_wait();
    __syncthreads();

    WaveState st;
#pragma unroll
    for (int i = 0; i < 16; ++i)
#pragma unroll
        for (int v = 0; v < 8; ++v) st.acc[i][v] = 0.0f;
    st.m = -1e30f;
    st.l = 0.0f;
    const int tql = wave * 16 + l16;

    constexpr int NIT = Te / BK;
    for (int j = 0; j < NIT; ++j) {
        const _Float16* er = erow0 + (j & 1) * LDS_R;
        const _Float16* et = etr0  + (j & 1) * LDS_T;

        if (j + 1 < NIT)   // overlap: copy tile j+1 while computing tile j
            stage_tile_async(enchB, enctB,
                             erow0 + ((j + 1) & 1) * LDS_R,
                             etr0  + ((j + 1) & 1) * LDS_T,
                             (j + 1) * BK, tid);

        attn_tile(st, q_lds, er, et, l16, half, tql);

        cp_wait();         // tile j+1 landed
        __syncthreads();   // everyone done reading tile j / writing tile j+1
    }

    write_ctx(st, outB, t0 + tql, half);
}

// ---------------------------------------------------------------------------
// Slow path: self-contained (converts f32->f16 + transposes inline, per tile).
// ---------------------------------------------------------------------------
__global__ void __launch_bounds__(128)
luong_attn_slow(const float* __restrict__ enc,
                const float* __restrict__ dec,
                float* __restrict__ out)
{
    extern __shared__ _Float16 smem[];
    _Float16* q_lds = smem;
    _Float16* erow  = smem + LDS_Q;
    _Float16* etr   = smem + LDS_Q + LDS_R;

    const int tid  = threadIdx.x;
    const int lane = tid & 31;
    const int wave = tid >> 5;
    const int half = lane >> 4;
    const int l16  = lane & 15;

    const int wg = blockIdx.x;
    const int b  = wg / (Td / BQ);
    const int t0 = (wg % (Td / BQ)) * BQ;

    const float* encB = enc + (size_t)b * Te * Dm;
    const float* decB = dec + (size_t)b * Td * Dm;
    float*       outB = out + (size_t)b * Td * (2 * Dm);

    stage_q_and_out(decB, outB, q_lds, t0, tid);

    WaveState st;
#pragma unroll
    for (int i = 0; i < 16; ++i)
#pragma unroll
        for (int v = 0; v < 8; ++v) st.acc[i][v] = 0.0f;
    st.m = -1e30f;
    st.l = 0.0f;
    const int tql = wave * 16 + l16;

    for (int j = 0; j < Te / BK; ++j) {
        const int e0 = j * BK;
        __syncthreads();
        {
            const int el = tid >> 2;
            const int d0 = (tid & 3) * 64;
            const float* src = encB + (size_t)(e0 + el) * Dm + d0;
#pragma unroll
            for (int i = 0; i < 16; ++i) {
                f32x4 v4 = ((const f32x4*)src)[i];
                const int d = d0 + i * 4;
                H4 pk;
                pk.h[0] = (_Float16)v4.x; pk.h[1] = (_Float16)v4.y;
                pk.h[2] = (_Float16)v4.z; pk.h[3] = (_Float16)v4.w;
                *(u32x2*)(erow + el * RP + d) = pk.u;
                etr[(d + 0) * TP + el] = pk.h[0];
                etr[(d + 1) * TP + el] = pk.h[1];
                etr[(d + 2) * TP + el] = pk.h[2];
                etr[(d + 3) * TP + el] = pk.h[3];
            }
            if (j + 1 < Te / BK)
                __builtin_prefetch(encB + (size_t)(e0 + BK + el) * Dm + d0, 0, 1);
        }
        __syncthreads();

        attn_tile(st, q_lds, erow, etr, l16, half, tql);
    }

    write_ctx(st, outB, t0 + tql, half);
}

// ---------------------------------------------------------------------------
extern "C" void kernel_launch(void* const* d_in, const int* in_sizes, int n_in,
                              void* d_out, int out_size, void* d_ws, size_t ws_size,
                              hipStream_t stream) {
    (void)in_sizes; (void)n_in; (void)out_size;
    const float* enc = (const float*)d_in[0];
    const float* dec = (const float*)d_in[1];
    float*       out = (float*)d_out;

    dim3 grid(Bb * (Td / BQ));   // 256 workgroups
    dim3 block(128);             // 4 waves (wave32)

    if (d_ws != nullptr && ws_size >= WS_NEEDED) {
        _Float16* ench = (_Float16*)d_ws;
        _Float16* enct = ench + (size_t)Bb * Te * Dm;
        dim3 cg(Dm / 64, Te / 64, Bb);
        convert_enc_kernel<<<cg, dim3(256), 0, stream>>>(enc, ench, enct);
        luong_attn_fast<<<grid, block, LDS_BYTES_FAST, stream>>>(ench, enct, dec, out);
    } else {
        luong_attn_slow<<<grid, block, LDS_BYTES_SLOW, stream>>>(enc, dec, out);
    }
}
